// JLoss_4733053960213
// MI455X (gfx1250) — compile-verified, matrix operand order checked by reading, not run on hardware
//
#include <hip/hip_runtime.h>
#include <hip/hip_bf16.h>

typedef __attribute__((ext_vector_type(16))) _Float16 v16h;
typedef __attribute__((ext_vector_type(8)))  float    v8f;

#define BATCH   8
#define NCH     32
#define HWTOT   147456          // 384*384
#define BLK     256             // 8 waves of 32
#define WAVES   8
#define CPW     8               // 32-pixel chunks per wave
#define BLKS_PB 72              // 72*8*8*32 = 147456 pixels per batch
// workspace layout (floats): T[8][32][32] then cnt[8][32]
#define T_ELEMS   (BATCH * NCH * NCH)
#define CNT_ELEMS (BATCH * NCH)
#define WS_ELEMS  (T_ELEMS + CNT_ELEMS)

__device__ inline v16h load16f_as_h(const float* __restrict__ p) {
    const float4* q = (const float4*)p;      // 16B aligned by construction
    v16h a;
#pragma unroll
    for (int j = 0; j < 4; ++j) {
        float4 f = q[j];
        a[4 * j + 0] = (_Float16)f.x;
        a[4 * j + 1] = (_Float16)f.y;
        a[4 * j + 2] = (_Float16)f.z;
        a[4 * j + 3] = (_Float16)f.w;
    }
    return a;
}

__global__ void zero_ws_kernel(float* ws, int n) {
    int i = blockIdx.x * blockDim.x + threadIdx.x;
    if (i < n) ws[i] = 0.0f;
}

// per-batch class histogram -> float counts
__global__ __launch_bounds__(BLK) void count_kernel(const int* __restrict__ target,
                                                    float* __restrict__ cnt) {
    __shared__ float bins[NCH];
    const int b = blockIdx.y;
    if (threadIdx.x < NCH) bins[threadIdx.x] = 0.0f;
    __syncthreads();
    const int* tg = target + (size_t)b * HWTOT + (size_t)blockIdx.x * 4096;
#pragma unroll 4
    for (int j = threadIdx.x; j < 4096; j += BLK)
        atomicAdd(&bins[tg[j] & (NCH - 1)], 1.0f);
    __syncthreads();
    if (threadIdx.x < NCH) atomicAdd(&cnt[b * NCH + threadIdx.x], bins[threadIdx.x]);
}

// T[b,i,k] = sum_p pred[b,i,p] * onehot(target[b,p]==k) via v_wmma_f32_16x16x32_f16
__global__ __launch_bounds__(BLK) void jstat_wmma_kernel(const float* __restrict__ pred,
                                                         const int*   __restrict__ target,
                                                         float* __restrict__ T) {
    __shared__ float accT[NCH * NCH];
    for (int t = threadIdx.x; t < NCH * NCH; t += BLK) accT[t] = 0.0f;
    __syncthreads();

    const int bb   = blockIdx.y;
    const int wave = threadIdx.x >> 5;
    const int lane = threadIdx.x & 31;
    const int half = lane >> 4;     // K-window selector (0: K 0-15, 1: K 16-31)
    const int mn   = lane & 15;     // A: row (channel), B: column (class)

    const int chunk0 = (blockIdx.x * WAVES + wave) * CPW;

    const float* predA0 = pred + (size_t)(bb * NCH + mn)      * HWTOT;
    const float* predA1 = pred + (size_t)(bb * NCH + 16 + mn) * HWTOT;
    const int*   tg     = target + (size_t)bb * HWTOT;

    v8f c00 = {}, c01 = {}, c10 = {}, c11 = {};

    for (int c = 0; c < CPW; ++c) {
        const int p0 = (chunk0 + c) * 32 + half * 16;   // 64B-aligned lane segment

        // A tiles: 16 consecutive fp32 -> f16 (channel mn and channel 16+mn)
        v16h a0 = load16f_as_h(predA0 + p0);
        v16h a1 = load16f_as_h(predA1 + p0);

        // Targets for this lane's K window (shared across lanes of the half -> L0 hit)
        int4 t0 = *(const int4*)(tg + p0);
        int4 t1 = *(const int4*)(tg + p0 + 4);
        int4 t2 = *(const int4*)(tg + p0 + 8);
        int4 t3 = *(const int4*)(tg + p0 + 12);
        int tt[16] = {t0.x, t0.y, t0.z, t0.w, t1.x, t1.y, t1.z, t1.w,
                      t2.x, t2.y, t2.z, t2.w, t3.x, t3.y, t3.z, t3.w};

        // B tiles: one-hot columns, classes mn and mn+16 (exact in f16)
        v16h b0, b1;
#pragma unroll
        for (int j = 0; j < 16; ++j) {
            b0[j] = (tt[j] == mn)        ? (_Float16)1.0f : (_Float16)0.0f;
            b1[j] = (tt[j] == (mn + 16)) ? (_Float16)1.0f : (_Float16)0.0f;
        }

        c00 = __builtin_amdgcn_wmma_f32_16x16x32_f16(false, a0, false, b0, (short)0, c00, false, false);
        c01 = __builtin_amdgcn_wmma_f32_16x16x32_f16(false, a0, false, b1, (short)0, c01, false, false);
        c10 = __builtin_amdgcn_wmma_f32_16x16x32_f16(false, a1, false, b0, (short)0, c10, false, false);
        c11 = __builtin_amdgcn_wmma_f32_16x16x32_f16(false, a1, false, b1, (short)0, c11, false, false);
    }

    // C/D layout: VGPR r, lane -> element (r + 8*half, mn). Reduce waves via LDS atomics.
#pragma unroll
    for (int r = 0; r < 8; ++r) {
        const int m = r + 8 * half;
        atomicAdd(&accT[(m)      * NCH + mn],      c00[r]);
        atomicAdd(&accT[(m)      * NCH + 16 + mn], c01[r]);
        atomicAdd(&accT[(16 + m) * NCH + mn],      c10[r]);
        atomicAdd(&accT[(16 + m) * NCH + 16 + mn], c11[r]);
    }
    __syncthreads();

    float* Tb = T + (size_t)bb * (NCH * NCH);
    for (int t = threadIdx.x; t < NCH * NCH; t += BLK) atomicAdd(&Tb[t], accT[t]);
}

// j[b] = -sum_{i!=k} log(0.5 + 0.5*(S_ii - S_ik)),  S_ik = T[b,i,k]/cnt[b,k]
__global__ __launch_bounds__(BLK) void finalize_kernel(const float* __restrict__ T,
                                                       const float* __restrict__ cnt,
                                                       float* __restrict__ out) {
    __shared__ float red[BLK];
    const int b = blockIdx.x;
    const float* Tb = T + (size_t)b * (NCH * NCH);
    const float* cb = cnt + (size_t)b * NCH;

    float s = 0.0f;
    for (int idx = threadIdx.x; idx < NCH * NCH; idx += BLK) {
        const int i = idx >> 5, k = idx & 31;
        if (i != k) {
            const float Sik = Tb[idx] / cb[k];
            const float Sii = Tb[i * NCH + i] / cb[i];
            s += __logf(0.5f + 0.5f * (Sii - Sik));
        }
    }
    red[threadIdx.x] = s;
    __syncthreads();
#pragma unroll
    for (int off = BLK / 2; off > 0; off >>= 1) {
        if (threadIdx.x < off) red[threadIdx.x] += red[threadIdx.x + off];
        __syncthreads();
    }
    if (threadIdx.x == 0) out[b] = -red[0];
}

extern "C" void kernel_launch(void* const* d_in, const int* in_sizes, int n_in,
                              void* d_out, int out_size, void* d_ws, size_t ws_size,
                              hipStream_t stream) {
    const float* pred   = (const float*)d_in[0];  // (8,32,384,384) fp32
    const int*   target = (const int*)d_in[1];    // (8,384,384) int
    float* out = (float*)d_out;                   // (8,) fp32
    float* T   = (float*)d_ws;                    // [8][32][32]
    float* cnt = T + T_ELEMS;                     // [8][32]

    zero_ws_kernel<<<(WS_ELEMS + BLK - 1) / BLK, BLK, 0, stream>>>((float*)d_ws, WS_ELEMS);
    count_kernel<<<dim3(HWTOT / 4096, BATCH), BLK, 0, stream>>>(target, cnt);
    jstat_wmma_kernel<<<dim3(BLKS_PB, BATCH), BLK, 0, stream>>>(pred, target, T);
    finalize_kernel<<<BATCH, BLK, 0, stream>>>(T, cnt, out);
    (void)in_sizes; (void)n_in; (void)out_size; (void)ws_size;
}
